// RelationalMessagePassingModule_85727547228489
// MI455X (gfx1250) — matrix operand order repaired and checked by reading, not compiled, not run on hardware
//
#include <hip/hip_runtime.h>

typedef __bf16 bf16_t;
typedef __attribute__((ext_vector_type(16))) __bf16 v16bf;
typedef __attribute__((ext_vector_type(8)))  __bf16 v8bf;
typedef __attribute__((ext_vector_type(4)))  __bf16 v4bf;
typedef __attribute__((ext_vector_type(8)))  float  v8f;
typedef __attribute__((ext_vector_type(4)))  float  v4f;

#define E 128

__device__ __forceinline__ v8f wmma_bf16(v16bf a, v16bf b, v8f c) {
    return __builtin_amdgcn_wmma_f32_16x16x32_bf16(false, a, false, b, (short)0, c, false, false);
}

// A-fragment (16x32 bf16) from an LDS tile stored row-major [16][ldk].
__device__ __forceinline__ v16bf load_a_frag(const bf16_t* tile, int ldk, int k0, int lane) {
    int row = lane & 15;
    int kb  = (lane >> 4) << 4;            // 0 or 16
    const bf16_t* p = tile + row * ldk + k0 + kb;
    union { v16bf v; v8bf h[2]; } u;
    u.h[0] = *(const v8bf*)(p);            // K = k0+kb .. +7
    u.h[1] = *(const v8bf*)(p + 8);        // K = k0+kb+8 .. +15
    return u.v;
}

// B-fragment (32x16 bf16) from a transposed weight matrix wT[Ncols][K] in global.
__device__ __forceinline__ v16bf load_b_frag(const bf16_t* __restrict__ wT, int K,
                                             int n0, int k0, int lane) {
    int col = lane & 15;
    int kb  = (lane >> 4) << 4;            // 0 or 16
    const bf16_t* p = wT + (size_t)(n0 + col) * K + k0 + kb;
    return *(const v16bf*)p;               // contiguous 32B
}

__device__ __forceinline__ void wait_lds() {
    asm volatile("s_wait_dscnt 0" ::: "memory");
}
__device__ __forceinline__ void wait_async() {
    asm volatile("s_wait_asynccnt 0" ::: "memory");
}
// CDNA5 async memory->LDS copy, 16B per lane, tracked by ASYNCcnt.
__device__ __forceinline__ void async_gather_b128(unsigned lds_addr, const void* gptr) {
    asm volatile("global_load_async_to_lds_b128 %0, %1, off"
                 :: "v"(lds_addr), "v"(gptr) : "memory");
}

// ---------------------------------------------------------------------------
// Prep kernels
// ---------------------------------------------------------------------------
__global__ void zero_f32_kernel(float* __restrict__ p, long long n) {
    long long i = (long long)blockIdx.x * blockDim.x + threadIdx.x;
    long long s = (long long)gridDim.x * blockDim.x;
    for (; i < n; i += s) p[i] = 0.0f;
}

__global__ void cvt_bf16_kernel(const float* __restrict__ in, bf16_t* __restrict__ out,
                                long long n) {
    long long i = (long long)blockIdx.x * blockDim.x + threadIdx.x;
    long long s = (long long)gridDim.x * blockDim.x;
    for (; i < n; i += s) out[i] = (bf16_t)in[i];
}

// in: fp32 [K][Ncols] row-major  ->  out: bf16 [Ncols][K]
__global__ void cvt_transpose_kernel(const float* __restrict__ in, bf16_t* __restrict__ out,
                                     int K, int Ncols) {
    int i = blockIdx.x * blockDim.x + threadIdx.x;
    if (i < K * Ncols) {
        int k = i / Ncols;
        int n = i - k * Ncols;
        out[(size_t)n * K + k] = (bf16_t)in[i];
    }
}

// ---------------------------------------------------------------------------
// Relation message kernel: msg = t + MLP(t), scatter-add into agg (atomics)
//   embbf: bf16 copy of node embeddings [Nn][128]
//   W1T: [E][ARITY*E] bf16 (transposed), W2T: [ARITY*E][E] bf16 (transposed)
// One wave handles 16 tuples; waves are fully independent.
// ---------------------------------------------------------------------------
template <int ARITY>
__global__ __launch_bounds__(128) void relation_kernel(
    const bf16_t* __restrict__ embbf, const int* __restrict__ idx,
    const bf16_t* __restrict__ W1T, const float* __restrict__ b1,
    const bf16_t* __restrict__ W2T, const float* __restrict__ b2,
    float* __restrict__ agg, int T) {

    constexpr int K1 = ARITY * E;                      // layer-1 K (and layer-2 N)
    __shared__ __attribute__((aligned(32))) bf16_t Atile[4][16 * K1];
    __shared__ __attribute__((aligned(32))) bf16_t Htile[4][16 * E];

    const int wave = threadIdx.x >> 5;
    const int lane = threadIdx.x & 31;
    const int tuple0 = (blockIdx.x * 4 + wave) * 16;
    if (tuple0 >= T) return;

    bf16_t* At = Atile[wave];
    bf16_t* Ht = Htile[wave];

    // ---- async gather: 16*ARITY rows of 256B each, 2 rows per instruction ----
    // At viewed flat as [16*ARITY][128] bf16 rows (matches [16][ARITY*E]).
    {
        const int sub = lane >> 4;                     // which of the 2 rows
        const int bo  = (lane & 15) * 16;              // byte offset within row
        const unsigned lds_base = (unsigned)(size_t)At;
        #pragma unroll
        for (int c = 0; c < 16 * ARITY; c += 2) {
            int node = idx[(size_t)tuple0 * ARITY + c + sub];
            const char* g = (const char*)embbf + (size_t)node * 256 + bo;
            async_gather_b128(lds_base + (unsigned)(c + sub) * 256 + bo, g);
        }
    }
    wait_async();

    const int nl = lane & 15;
    const int mb = (lane >> 4) << 3;                   // row base 0 or 8

    // ---- layer 1: h = relu(t @ W1 + b1), H = 128 ----
    v8f acc[8] = {};
    for (int k0 = 0; k0 < K1; k0 += 32) {
        v16bf a = load_a_frag(At, K1, k0, lane);
        #pragma unroll
        for (int n = 0; n < 8; ++n) {
            v16bf b = load_b_frag(W1T, K1, n * 16, k0, lane);
            acc[n] = wmma_bf16(a, b, acc[n]);
        }
    }
    #pragma unroll
    for (int n = 0; n < 8; ++n) {
        float bias = b1[n * 16 + nl];
        #pragma unroll
        for (int r = 0; r < 8; ++r) {
            float v = acc[n][r] + bias;
            v = v > 0.0f ? v : 0.0f;
            Ht[(mb + r) * E + n * 16 + nl] = (bf16_t)v;
        }
    }
    wait_lds();

    // ---- layer 2: m = h @ W2 + b2 ; msg = t + m ; scatter-add ----
    #pragma unroll
    for (int ha = 0; ha < ARITY; ++ha) {               // 128-column halves of N2
        v8f acc2[8] = {};
        for (int k0 = 0; k0 < E; k0 += 32) {
            v16bf a = load_a_frag(Ht, E, k0, lane);
            #pragma unroll
            for (int n = 0; n < 8; ++n) {
                v16bf b = load_b_frag(W2T, E, ha * E + n * 16, k0, lane);
                acc2[n] = wmma_bf16(a, b, acc2[n]);
            }
        }
        // destination nodes for the 8 rows this lane touches
        int nodes[8];
        #pragma unroll
        for (int r = 0; r < 8; ++r)
            nodes[r] = idx[(size_t)(tuple0 + mb + r) * ARITY + ha];

        #pragma unroll
        for (int n = 0; n < 8; ++n) {
            int c = ha * E + n * 16 + nl;              // column in t-space
            float bias = b2[c];
            #pragma unroll
            for (int r = 0; r < 8; ++r) {
                float t = (float)At[(mb + r) * K1 + c];
                float v = acc2[n][r] + bias + t;
                atomicAdd(&agg[(size_t)nodes[r] * E + n * 16 + nl], v);
            }
        }
    }
}

// ---------------------------------------------------------------------------
// Update kernel: out = emb + MLP2([emb || agg])
// ---------------------------------------------------------------------------
__global__ __launch_bounds__(128) void update_kernel(
    const bf16_t* __restrict__ embbf, const float* __restrict__ emb,
    const float* __restrict__ agg,
    const bf16_t* __restrict__ W1T, const float* __restrict__ b1,
    const bf16_t* __restrict__ W2T, const float* __restrict__ b2,
    float* __restrict__ out, int Nn) {

    constexpr int K1 = 2 * E;
    __shared__ __attribute__((aligned(32))) bf16_t Atile[4][16 * K1];
    __shared__ __attribute__((aligned(32))) bf16_t Htile[4][16 * E];

    const int wave = threadIdx.x >> 5;
    const int lane = threadIdx.x & 31;
    const int row0 = (blockIdx.x * 4 + wave) * 16;
    if (row0 >= Nn) return;

    bf16_t* At = Atile[wave];
    bf16_t* Ht = Htile[wave];

    // emb half: async bf16 copy (row stride in LDS = 512B, emb occupies first 256B)
    {
        const int sub = lane >> 4;
        const int bo  = (lane & 15) * 16;
        const unsigned lds_base = (unsigned)(size_t)At;
        #pragma unroll
        for (int c = 0; c < 16; c += 2) {
            const char* g = (const char*)embbf + (size_t)(row0 + c + sub) * 256 + bo;
            async_gather_b128(lds_base + (unsigned)(c + sub) * 512 + bo, g);
        }
    }
    // agg half: fp32 load + convert
    for (int r = 0; r < 16; ++r) {
        v4f g = *(const v4f*)(agg + (size_t)(row0 + r) * E + lane * 4);
        v4bf tg;
        tg[0]=(bf16_t)g[0]; tg[1]=(bf16_t)g[1]; tg[2]=(bf16_t)g[2]; tg[3]=(bf16_t)g[3];
        *(v4bf*)(At + r * K1 + E + lane * 4) = tg;
    }
    wait_async();
    wait_lds();

    const int nl = lane & 15;
    const int mb = (lane >> 4) << 3;

    v8f acc[8] = {};
    for (int k0 = 0; k0 < K1; k0 += 32) {
        v16bf a = load_a_frag(At, K1, k0, lane);
        #pragma unroll
        for (int n = 0; n < 8; ++n) {
            v16bf b = load_b_frag(W1T, K1, n * 16, k0, lane);
            acc[n] = wmma_bf16(a, b, acc[n]);
        }
    }
    #pragma unroll
    for (int n = 0; n < 8; ++n) {
        float bias = b1[n * 16 + nl];
        #pragma unroll
        for (int r = 0; r < 8; ++r) {
            float v = acc[n][r] + bias;
            v = v > 0.0f ? v : 0.0f;
            Ht[(mb + r) * E + n * 16 + nl] = (bf16_t)v;
        }
    }
    wait_lds();

    v8f acc2[8] = {};
    for (int k0 = 0; k0 < E; k0 += 32) {
        v16bf a = load_a_frag(Ht, E, k0, lane);
        #pragma unroll
        for (int n = 0; n < 8; ++n) {
            v16bf b = load_b_frag(W2T, E, n * 16, k0, lane);
            acc2[n] = wmma_bf16(a, b, acc2[n]);
        }
    }
    #pragma unroll
    for (int n = 0; n < 8; ++n) {
        int c = n * 16 + nl;
        float bias = b2[c];
        #pragma unroll
        for (int r = 0; r < 8; ++r) {
            size_t o = (size_t)(row0 + mb + r) * E + c;
            out[o] = emb[o] + acc2[n][r] + bias;       // fp32 residual from global
        }
    }
}

// ---------------------------------------------------------------------------
extern "C" void kernel_launch(void* const* d_in, const int* in_sizes, int n_in,
                              void* d_out, int out_size, void* d_ws, size_t ws_size,
                              hipStream_t stream) {
    const float* emb = (const float*)d_in[0];
    const int*   ia  = (const int*)d_in[1];
    const int*   ib  = (const int*)d_in[2];
    const int*   ic  = (const int*)d_in[3];
    const float* Wa1 = (const float*)d_in[4];  const float* ba1 = (const float*)d_in[5];
    const float* Wa2 = (const float*)d_in[6];  const float* ba2 = (const float*)d_in[7];
    const float* Wb1 = (const float*)d_in[8];  const float* bb1 = (const float*)d_in[9];
    const float* Wb2 = (const float*)d_in[10]; const float* bb2 = (const float*)d_in[11];
    const float* Wc1 = (const float*)d_in[12]; const float* bc1 = (const float*)d_in[13];
    const float* Wc2 = (const float*)d_in[14]; const float* bc2 = (const float*)d_in[15];
    const float* Wu1 = (const float*)d_in[16]; const float* bu1 = (const float*)d_in[17];
    const float* Wu2 = (const float*)d_in[18]; const float* bu2 = (const float*)d_in[19];

    const int Nn  = in_sizes[0] / E;     // 100000
    const int Ta  = in_sizes[1] / 2;     // 250000 binary tuples
    const int Tb  = in_sizes[2] / 2;
    const int Tc  = in_sizes[3];         // 200000 unary tuples

    // workspace layout
    char* ws = (char*)d_ws;
    float*  agg   = (float*)ws;
    size_t  off   = (size_t)Nn * E * sizeof(float);
    bf16_t* embbf = (bf16_t*)(ws + off); off += (size_t)Nn * E * 2;
    bf16_t* Wa1T  = (bf16_t*)(ws + off); off += (size_t)256 * 128 * 2;
    bf16_t* Wa2T  = (bf16_t*)(ws + off); off += (size_t)128 * 256 * 2;
    bf16_t* Wb1T  = (bf16_t*)(ws + off); off += (size_t)256 * 128 * 2;
    bf16_t* Wb2T  = (bf16_t*)(ws + off); off += (size_t)128 * 256 * 2;
    bf16_t* Wc1T  = (bf16_t*)(ws + off); off += (size_t)128 * 128 * 2;
    bf16_t* Wc2T  = (bf16_t*)(ws + off); off += (size_t)128 * 128 * 2;
    bf16_t* Wu1T  = (bf16_t*)(ws + off); off += (size_t)256 * 128 * 2;
    bf16_t* Wu2T  = (bf16_t*)(ws + off); off += (size_t)128 * 128 * 2;

    // prep: zero agg, bf16 embedding table, convert/transpose weights
    zero_f32_kernel<<<4096, 256, 0, stream>>>(agg, (long long)Nn * E);
    cvt_bf16_kernel<<<4096, 256, 0, stream>>>(emb, embbf, (long long)Nn * E);
    cvt_transpose_kernel<<<(256*128 + 255)/256, 256, 0, stream>>>(Wa1, Wa1T, 256, 128);
    cvt_transpose_kernel<<<(128*256 + 255)/256, 256, 0, stream>>>(Wa2, Wa2T, 128, 256);
    cvt_transpose_kernel<<<(256*128 + 255)/256, 256, 0, stream>>>(Wb1, Wb1T, 256, 128);
    cvt_transpose_kernel<<<(128*256 + 255)/256, 256, 0, stream>>>(Wb2, Wb2T, 128, 256);
    cvt_transpose_kernel<<<(128*128 + 255)/256, 256, 0, stream>>>(Wc1, Wc1T, 128, 128);
    cvt_transpose_kernel<<<(128*128 + 255)/256, 256, 0, stream>>>(Wc2, Wc2T, 128, 128);
    cvt_transpose_kernel<<<(256*128 + 255)/256, 256, 0, stream>>>(Wu1, Wu1T, 256, 128);
    cvt_transpose_kernel<<<(128*128 + 255)/256, 256, 0, stream>>>(Wu2, Wu2T, 128, 128);

    // relation message passes (scatter-add into agg)
    int blkA = (Ta / 16 + 3) / 4;
    int blkB = (Tb / 16 + 3) / 4;
    int blkC = (Tc / 16 + 3) / 4;
    relation_kernel<2><<<blkA, 128, 0, stream>>>(embbf, ia, Wa1T, ba1, Wa2T, ba2, agg, Ta);
    relation_kernel<2><<<blkB, 128, 0, stream>>>(embbf, ib, Wb1T, bb1, Wb2T, bb2, agg, Tb);
    relation_kernel<1><<<blkC, 128, 0, stream>>>(embbf, ic, Wc1T, bc1, Wc2T, bc2, agg, Tc);

    // node update
    int blkU = (Nn / 16 + 3) / 4;
    update_kernel<<<blkU, 128, 0, stream>>>(embbf, emb, agg, Wu1T, bu1, Wu2T, bu2,
                                            (float*)d_out, Nn);
}
